// EdgePredictor_34497177321789
// MI455X (gfx1250) — matrix operand order repaired and checked by reading, not compiled, not run on hardware
//
#include <hip/hip_runtime.h>
#include <hip/hip_bf16.h>
#include <math.h>

typedef __attribute__((ext_vector_type(16))) _Float16 v16h;
typedef __attribute__((ext_vector_type(8)))  _Float16 v8h;
typedef __attribute__((ext_vector_type(8)))  float    v8f;

#define BN_EPS 1e-5f

// -------------------------------------------------------------------------
// Phase 1a: A2[b][i][c] = s1[c]*(W1[:, :128] @ x[b,:,i])[c] + (s1*(b1-m1)+be1)[c]
//           Bp[b][j][c] = s1[c]*(W1[:,128:] @ x[b,:,j])[c]
// Layout [b][point][channel] so the fused kernel reads channel-contiguous.
// -------------------------------------------------------------------------
__global__ void precompute_ab_kernel(const float* __restrict__ x,
                                     const float* __restrict__ W1,
                                     const float* __restrict__ b1,
                                     const float* __restrict__ g1,
                                     const float* __restrict__ be1,
                                     const float* __restrict__ m1,
                                     const float* __restrict__ v1,
                                     float* __restrict__ A2,
                                     float* __restrict__ Bp)
{
    int idx = blockIdx.x * 256 + threadIdx.x;          // 8*256*512 threads
    if (idx >= 8 * 256 * 512) return;
    int c = idx & 511;
    int i = (idx >> 9) & 255;
    int b = idx >> 17;
    const float* xc = x + b * 128 * 256 + i;           // x[b][k][i], stride 256
    const float* w  = W1 + c * 256;
    float da = 0.f, db = 0.f;
#pragma unroll 8
    for (int k = 0; k < 128; ++k) {
        float xv = xc[k * 256];
        da += w[k]       * xv;
        db += w[128 + k] * xv;
    }
    float s = g1[c] * rsqrtf(v1[c] + BN_EPS);
    A2[idx] = s * da + (s * (b1[c] - m1[c]) + be1[c]);
    Bp[idx] = s * db;
}

// -------------------------------------------------------------------------
// Phase 1b: fold BN scales into f16 weights + bias constants
//   W2h = f16(diag(s2) W2)   c2 = s2*(b2-m2)+be2
//   W3h = f16(diag(s3) W3)   c3 = s3*(b3-m3)+be3
// -------------------------------------------------------------------------
__global__ void prep_weights_kernel(const float* __restrict__ W2, const float* __restrict__ b2,
                                    const float* __restrict__ g2, const float* __restrict__ be2,
                                    const float* __restrict__ m2, const float* __restrict__ v2,
                                    const float* __restrict__ W3, const float* __restrict__ b3,
                                    const float* __restrict__ g3, const float* __restrict__ be3,
                                    const float* __restrict__ m3, const float* __restrict__ v3,
                                    _Float16* __restrict__ W2h, float* __restrict__ c2o,
                                    _Float16* __restrict__ W3h, float* __restrict__ c3o)
{
    int idx = blockIdx.x * 256 + threadIdx.x;
    if (idx < 131072) {                                // W2: 256x512
        int oc = idx >> 9;
        float s = g2[oc] * rsqrtf(v2[oc] + BN_EPS);
        W2h[idx] = (_Float16)(s * W2[idx]);
    } else if (idx < 131072 + 16384) {                 // W3: 64x256
        int t  = idx - 131072;
        int oc = t >> 8;
        float s = g3[oc] * rsqrtf(v3[oc] + BN_EPS);
        W3h[t] = (_Float16)(s * W3[t]);
    } else if (idx < 147456 + 256) {                   // c2
        int oc = idx - 147456;
        float s = g2[oc] * rsqrtf(v2[oc] + BN_EPS);
        c2o[oc] = s * (b2[oc] - m2[oc]) + be2[oc];
    } else if (idx < 147712 + 64) {                    // c3
        int oc = idx - 147712;
        float s = g3[oc] * rsqrtf(v3[oc] + BN_EPS);
        c3o[oc] = s * (b3[oc] - m3[oc]) + be3[oc];
    }
}

// -------------------------------------------------------------------------
// Fused MLP over a (batch b, row i, 32 consecutive j) tile.
// 256 threads = 8 wave32; all matmuls via v_wmma_f32_16x16x32_f16.
//
// WMMA fragment mapping (CDNA5 ISA 7.12.2, wave32):
//   A (16x32 f16, MxK): lane L holds M = L&15; element e of v16h holds
//       K = (L>>4)*8 + e          for e in 0..7
//       K = 16 + (L>>4)*8 + (e-8) for e in 8..15
//   B (32x16 f16, KxN): lane L holds N = L&15; element e holds the same K as in A
//   C/D (16x16 f32):    lane L holds N = L&15; VGPR v holds M = (L>>4)*8 + v
// h1/h2 are stored [point][channel] in LDS so each lane's B-fragment is two
// 16-byte reads (2 x ds_load_b128); W rows give two 16-byte global
// loads per A-fragment.
// -------------------------------------------------------------------------
__global__ __launch_bounds__(256)
void edge_mlp_wmma_kernel(const float* __restrict__ A2, const float* __restrict__ Bp,
                          const _Float16* __restrict__ W2h, const float* __restrict__ c2,
                          const _Float16* __restrict__ W3h, const float* __restrict__ c3,
                          const float* __restrict__ W4, const float* __restrict__ b4,
                          float* __restrict__ out)
{
    __shared__ _Float16 h1T[32][520];   // 512 ch + 8 pad   (33280 B)
    __shared__ _Float16 h2T[32][264];   // 256 ch + 8 pad   (16896 B)
    __shared__ float    h3T[32][68];    //  64 ch + 4 pad   ( 8704 B)

    int bid = blockIdx.x;               // 16384 blocks: b(8) x i(256) x jblk(8)
    int b   = bid >> 11;
    int rem = bid & 2047;
    int i   = rem >> 3;
    int j0  = (rem & 7) << 5;

    int tid  = threadIdx.x;
    int lane = tid & 31;
    int wave = tid >> 5;
    int lq   = lane & 15;               // M/N coordinate within tile
    int lh   = lane >> 4;               // K-half / M-half select

    union frag { v16h v; v8h h[2]; };

    // ---- Stage A: h1 = relu(A[:,i] + B[:,j]), stored transposed in f16 ----
    const float* Arow  = A2 + (((size_t)b << 8) + i) * 512;
    const float* Bbase = Bp + (((size_t)b << 8) + j0) * 512;
    for (int e = tid; e < 32 * 512; e += 256) {
        int p = e >> 9, c = e & 511;
        float v = Arow[c] + Bbase[p * 512 + c];
        h1T[p][c] = (_Float16)fmaxf(v, 0.f);
    }
    __syncthreads();

    // ---- Stage B: h2[256][32] = relu(W2h @ h1 + c2), K = 512 ----
    for (int r = 0; r < 4; ++r) {
        int t  = wave + (r << 3);       // 0..31 output tiles
        int mt = t >> 1, nt = t & 1;
        v8f acc = {};
        const _Float16* wrow = W2h + (size_t)(mt * 16 + lq) * 512 + lh * 8;
        const _Float16* hrow = &h1T[nt * 16 + lq][lh * 8];
        for (int kk = 0; kk < 16; ++kk) {
            int k0 = kk << 5;
            frag a, bm;
            a.h[0]  = *(const v8h*)(wrow + k0);        // K = k0+base .. +7
            a.h[1]  = *(const v8h*)(wrow + k0 + 16);   // K = k0+16+base .. +7
            bm.h[0] = *(const v8h*)(hrow + k0);        // K = k0+half*8 .. +7
            bm.h[1] = *(const v8h*)(hrow + k0 + 16);   // K = k0+16+half*8 .. +7
            acc = __builtin_amdgcn_wmma_f32_16x16x32_f16(false, a.v, false, bm.v,
                                                         (short)0, acc, false, false);
        }
        int mb = mt * 16 + lh * 8;
        int n  = nt * 16 + lq;
        v8h hv;
#pragma unroll
        for (int q = 0; q < 8; ++q)
            hv[q] = (_Float16)fmaxf(acc[q] + c2[mb + q], 0.f);
        *(v8h*)&h2T[n][mb] = hv;        // 8 consecutive channels: one b128 store
    }
    __syncthreads();

    // ---- Stage C: h3[64][32] = relu(W3h @ h2 + c3), K = 256 ----
    {
        int mt = wave >> 1, nt = wave & 1;   // 8 tiles, 1 per wave
        v8f acc = {};
        const _Float16* wrow = W3h + (size_t)(mt * 16 + lq) * 256 + lh * 8;
        const _Float16* hrow = &h2T[nt * 16 + lq][lh * 8];
        for (int kk = 0; kk < 8; ++kk) {
            int k0 = kk << 5;
            frag a, bm;
            a.h[0]  = *(const v8h*)(wrow + k0);
            a.h[1]  = *(const v8h*)(wrow + k0 + 16);
            bm.h[0] = *(const v8h*)(hrow + k0);
            bm.h[1] = *(const v8h*)(hrow + k0 + 16);
            acc = __builtin_amdgcn_wmma_f32_16x16x32_f16(false, a.v, false, bm.v,
                                                         (short)0, acc, false, false);
        }
        int mb = mt * 16 + lh * 8;
        int n  = nt * 16 + lq;
#pragma unroll
        for (int q = 0; q < 8; ++q)
            h3T[n][mb + q] = fmaxf(acc[q] + c3[mb + q], 0.f);
    }
    __syncthreads();

    // ---- Stage D: out = sigmoid(W4 . h3 + b4) ----
    if (tid < 32) {
        float acc = b4[0];
#pragma unroll 8
        for (int m = 0; m < 64; ++m)
            acc += W4[m] * h3T[tid][m];
        out[(size_t)b * 65536 + (size_t)i * 256 + j0 + tid] = 1.f / (1.f + expf(-acc));
    }
}

// -------------------------------------------------------------------------
extern "C" void kernel_launch(void* const* d_in, const int* in_sizes, int n_in,
                              void* d_out, int out_size, void* d_ws, size_t ws_size,
                              hipStream_t stream)
{
    (void)in_sizes; (void)n_in; (void)out_size; (void)ws_size;
    const float* x   = (const float*)d_in[0];
    const float* W1  = (const float*)d_in[1];
    const float* b1  = (const float*)d_in[2];
    const float* g1  = (const float*)d_in[3];
    const float* be1 = (const float*)d_in[4];
    const float* m1  = (const float*)d_in[5];
    const float* v1  = (const float*)d_in[6];
    const float* W2  = (const float*)d_in[7];
    const float* b2  = (const float*)d_in[8];
    const float* g2  = (const float*)d_in[9];
    const float* be2 = (const float*)d_in[10];
    const float* m2  = (const float*)d_in[11];
    const float* v2  = (const float*)d_in[12];
    const float* W3  = (const float*)d_in[13];
    const float* b3  = (const float*)d_in[14];
    const float* g3  = (const float*)d_in[15];
    const float* be3 = (const float*)d_in[16];
    const float* m3  = (const float*)d_in[17];
    const float* v3  = (const float*)d_in[18];
    const float* W4  = (const float*)d_in[19];
    const float* b4  = (const float*)d_in[20];
    float* out = (float*)d_out;

    // Workspace layout (~8.6 MB)
    char* ws = (char*)d_ws;
    float*    A2  = (float*)(ws);                                   // 4 MiB
    float*    Bp  = (float*)(ws + (4u << 20));                      // 4 MiB
    _Float16* W2h = (_Float16*)(ws + (8u << 20));                   // 256 KiB
    _Float16* W3h = (_Float16*)(ws + (8u << 20) + (256u << 10));    // 32 KiB
    float*    c2  = (float*)(ws + (8u << 20) + (292u << 10));       // 1 KiB
    float*    c3  = c2 + 256;

    precompute_ab_kernel<<<(8 * 256 * 512) / 256, 256, 0, stream>>>(
        x, W1, b1, g1, be1, m1, v1, A2, Bp);
    prep_weights_kernel<<<(147776 + 255) / 256, 256, 0, stream>>>(
        W2, b2, g2, be2, m2, v2, W3, b3, g3, be3, m3, v3, W2h, c2, W3h, c3);
    edge_mlp_wmma_kernel<<<16384, 256, 0, stream>>>(
        A2, Bp, W2h, c2, W3h, c3, W4, b4, out);
}
